// SelfAttnLayer_81870666596627
// MI455X (gfx1250) — matrix operand order repaired and checked by reading, not compiled, not run on hardware
//
#include <hip/hip_runtime.h>
#include <hip/hip_bf16.h>

typedef __attribute__((ext_vector_type(16))) _Float16 v16h;
typedef __attribute__((ext_vector_type(8)))  float    v8f;
typedef __attribute__((ext_vector_type(2)))  _Float16 v2h;
typedef __attribute__((ext_vector_type(2)))  __fp16   v2fp16;

#define HW   4096
#define NCH  64
#define DQK  8

// sqrt(log2(e)): folded into Q (and therefore K, since K==Q) so the WMMA
// scores come out pre-multiplied by log2(e) and softmax can use raw exp2.
#define QSCALE 1.2011224087f

union V16 {
    v16h v;
    uint4 u[2];
    uint2 d[4];
    unsigned w[8];
    _Float16 h[16];
};

union H2 { v2h h2; _Float16 h[2]; unsigned u; };

#if __has_builtin(__builtin_amdgcn_exp2f)
#define EXP2F(x) __builtin_amdgcn_exp2f(x)
#else
#define EXP2F(x) exp2f(x)
#endif

// one v_cvt_pk_rtz_f16_f32: pack two f32 into f16x2 (low = a, high = b)
__device__ __forceinline__ unsigned pack2(float a, float b) {
    union { v2fp16 h2; unsigned u; } t;
    t.h2 = __builtin_amdgcn_cvt_pkrtz(a, b);
    return t.u;
}

// packed f16 max (V_PK_MAX_NUM_F16): non-volatile asm -> schedulable, and no
// backend-inserted canonicalize (unlike __builtin_elementwise_max on bitcasts)
__device__ __forceinline__ unsigned pkmax(unsigned a, unsigned b) {
    unsigned d;
    asm("v_pk_max_num_f16 %0, %1, %2" : "=v"(d) : "v"(a), "v"(b));
    return d;
}

// xor-butterfly within each 16-lane half, pure VALU via v_permlane16_b32
template <int STEP>
__device__ __forceinline__ unsigned xor16(unsigned v) {
#if __has_builtin(__builtin_amdgcn_permlane16)
    constexpr unsigned LO[4] = {0x67452301u, 0x54761032u, 0x32107654u, 0xFEDCBA98u};
    constexpr unsigned HI[4] = {0xEFCDAB89u, 0xDCFE98BAu, 0xBA98FEDCu, 0x76543210u};
    return __builtin_amdgcn_permlane16(v, v, LO[STEP], HI[STEP], false, false);
#else
    return (unsigned)__shfl_xor((int)v, 1 << STEP);
#endif
}

// ---------------------------------------------------------------------------
// Pass 1: Q = (X*Wq + bq) * QSCALE -> f16 [B][HW][8]
//         V = X*Wv + bv           -> f16 TRANSPOSED [B][64ch][HW]
// ---------------------------------------------------------------------------
__global__ void qv_pack_kernel(const float* __restrict__ X,
                               const float* __restrict__ Wq,
                               const float* __restrict__ bq,
                               const float* __restrict__ Wv,
                               const float* __restrict__ bv,
                               _Float16* __restrict__ Qh,
                               _Float16* __restrict__ Vt) {
    __shared__ float Xs[64][NCH];
    const int b     = blockIdx.x >> 6;
    const int tile  = blockIdx.x & 63;
    const int nbase = tile * 64;
    const int t     = threadIdx.x;

    const float* Xbase = X + ((size_t)b * HW + nbase) * NCH;
#pragma unroll
    for (int i = 0; i < 16; ++i) {
        int idx = i * 256 + t;
        Xs[idx >> 6][idx & 63] = Xbase[idx];
    }
    __syncthreads();

    // ---- V: one channel per thread over 16 consecutive pixels ----
    const int c     = t & 63;
    const int pbase = (t >> 6) * 16;
    float acc[16];
    const float bvc = bv[c];
#pragma unroll
    for (int i = 0; i < 16; ++i) acc[i] = bvc;
    for (int k = 0; k < NCH; ++k) {
        float w = Wv[k * NCH + c];
#pragma unroll
        for (int i = 0; i < 16; ++i) acc[i] = fmaf(Xs[pbase + i][k], w, acc[i]);
    }
    _Float16* vdst = Vt + ((size_t)(b * NCH + c) * HW + nbase + pbase);
    V16 pk;
#pragma unroll
    for (int i = 0; i < 16; ++i) pk.h[i] = (_Float16)acc[i];
    *(uint4*)(vdst)     = pk.u[0];
    *(uint4*)(vdst + 8) = pk.u[1];

    // ---- Q (scaled): 512 outputs, 2 per thread ----
    const int qc = t & 7;
    const int p0 = t >> 3;
#pragma unroll
    for (int r = 0; r < 2; ++r) {
        int p = p0 + r * 32;
        float a = bq[qc];
        for (int k = 0; k < NCH; ++k) a = fmaf(Xs[p][k], Wq[k * DQK + qc], a);
        Qh[((size_t)b * HW + nbase + p) * DQK + qc] = (_Float16)(a * QSCALE);
    }
}

// ---------------------------------------------------------------------------
// Pass 2: streaming (flash) attention, one wave32 per 16-query tile.
// ---------------------------------------------------------------------------
__global__ void __launch_bounds__(128)
flash_attn_kernel(const _Float16* __restrict__ Qh,
                  const _Float16* __restrict__ Vt,
                  const float* __restrict__ X,
                  const float* __restrict__ gamma_p,
                  float* __restrict__ out) {
    const int widx  = threadIdx.x >> 5;
    const int lane  = threadIdx.x & 31;
    const int gw    = blockIdx.x * 4 + widx;
    const int b     = gw >> 8;
    const int qtile = gw & 255;
    const int qbase = qtile * 16;
    const int ln    = lane & 15;
    const bool hi   = lane >= 16;

    __shared__ _Float16 Pl[4][16 * 40];
    _Float16* pl = Pl[widx];

    // Q A-fragment: lanes 0-15 carry K=0..7 in v0..v3, everything else zero
    V16 qa = {};
    if (!hi) {
        qa.u[0] = *(const uint4*)(Qh + ((size_t)b * HW + qbase + ln) * DQK);
    }

    V16 ones;
#pragma unroll
    for (int i = 0; i < 16; ++i) ones.h[i] = (_Float16)1.0f;

    v8f acc0 = {0,0,0,0,0,0,0,0};
    v8f acc1 = {0,0,0,0,0,0,0,0};
    v8f acc2 = {0,0,0,0,0,0,0,0};
    v8f acc3 = {0,0,0,0,0,0,0,0};
    v8f lacc = {0,0,0,0,0,0,0,0};
    const v8f zero = {0,0,0,0,0,0,0,0};

    // running row max: packed f16 (4 regs) + f32 mirror used by exp2
    unsigned m_pk[4];
    float mf[8];
#pragma unroll
    for (int j = 0; j < 4; ++j) m_pk[j] = 0xFBFFFBFFu;     // f16 -65504 pair
#pragma unroll
    for (int i = 0; i < 8; ++i) mf[i] = -65504.0f;

    // K fragments: zero-init once, only u[0] of lanes 0-15 is live
    V16 kb0 = {}, kb1 = {};

    const _Float16* Kb = Qh + (size_t)b * HW * DQK;   // K matrix == Q matrix
    const _Float16* Vb = Vt + (size_t)b * NCH * HW;

    for (int kc = 0; kc < HW; kc += 32) {
        // ---- issue all global loads for this chunk up front ----
        if (!hi) {
            kb0.u[0] = *(const uint4*)(Kb + (size_t)(kc + ln) * DQK);
            kb1.u[0] = *(const uint4*)(Kb + (size_t)(kc + 16 + ln) * DQK);
        }
        V16 vf0, vf1, vf2, vf3;
        {
            const size_t off = (size_t)kc + (hi ? 8 : 0);
            const _Float16* vp0 = Vb + (size_t)(0 * 16 + ln) * HW + off;
            const _Float16* vp1 = Vb + (size_t)(1 * 16 + ln) * HW + off;
            const _Float16* vp2 = Vb + (size_t)(2 * 16 + ln) * HW + off;
            const _Float16* vp3 = Vb + (size_t)(3 * 16 + ln) * HW + off;
            vf0.u[0] = *(const uint4*)(vp0); vf0.u[1] = *(const uint4*)(vp0 + 16);
            vf1.u[0] = *(const uint4*)(vp1); vf1.u[1] = *(const uint4*)(vp1 + 16);
            vf2.u[0] = *(const uint4*)(vp2); vf2.u[1] = *(const uint4*)(vp2 + 16);
            vf3.u[0] = *(const uint4*)(vp3); vf3.u[1] = *(const uint4*)(vp3 + 16);
        }

        v8f s0 = __builtin_amdgcn_wmma_f32_16x16x32_f16(false, qa.v, false, kb0.v,
                                                        (short)0, zero, false, false);
        v8f s1 = __builtin_amdgcn_wmma_f32_16x16x32_f16(false, qa.v, false, kb1.v,
                                                        (short)0, zero, false, false);

        // ---- row max: merge tiles + pack rows to f16x2, VALU butterfly ----
        unsigned cm[4];
#pragma unroll
        for (int j = 0; j < 4; ++j)
            cm[j] = pkmax(pack2(s0[2 * j], s0[2 * j + 1]),
                          pack2(s1[2 * j], s1[2 * j + 1]));
#pragma unroll
        for (int j = 0; j < 4; ++j) cm[j] = pkmax(cm[j], xor16<0>(cm[j]));
#pragma unroll
        for (int j = 0; j < 4; ++j) cm[j] = pkmax(cm[j], xor16<1>(cm[j]));
#pragma unroll
        for (int j = 0; j < 4; ++j) cm[j] = pkmax(cm[j], xor16<2>(cm[j]));
#pragma unroll
        for (int j = 0; j < 4; ++j) cm[j] = pkmax(cm[j], xor16<3>(cm[j]));

        unsigned mn_pk[4], du = 0;
#pragma unroll
        for (int j = 0; j < 4; ++j) {
            mn_pk[j] = pkmax(m_pk[j], cm[j]);
            du |= mn_pk[j] ^ m_pk[j];
        }

        // ---- rescale only when some row max actually increased ----
        if (__any((int)(du != 0u))) {
#pragma unroll
            for (int j = 0; j < 4; ++j) {
                H2 t2; t2.u = mn_pk[j];
                float n0 = (float)t2.h[0];
                float n1 = (float)t2.h[1];
                float sc0 = EXP2F(mf[2 * j]     - n0);
                float sc1 = EXP2F(mf[2 * j + 1] - n1);
                acc0[2 * j] *= sc0; acc0[2 * j + 1] *= sc1;
                acc1[2 * j] *= sc0; acc1[2 * j + 1] *= sc1;
                acc2[2 * j] *= sc0; acc2[2 * j + 1] *= sc1;
                acc3[2 * j] *= sc0; acc3[2 * j + 1] *= sc1;
                lacc[2 * j] *= sc0; lacc[2 * j + 1] *= sc1;
                mf[2 * j] = n0; mf[2 * j + 1] = n1;
                m_pk[j] = mn_pk[j];
            }
        }

        // ---- P = exp2(S - m); pack pair (tile0,tile1) per b32 LDS store ----
        // interleaved storage: col' = 2*key + tile, row stride 40 halves
#pragma unroll
        for (int i = 0; i < 8; ++i) {
            float p0 = EXP2F(s0[i] - mf[i]);
            float p1 = EXP2F(s1[i] - mf[i]);
            int row = i + (hi ? 8 : 0);
            *(unsigned*)(pl + row * 40 + 2 * ln) = pack2(p0, p1);
        }

        // ---- reload P in A-fragment layout: 4x b64 + 2x v_perm each ----
        const int cb = hi ? 16 : 0;
        V16 pa;
#pragma unroll
        for (int j = 0; j < 4; ++j) {
            uint2 dj = *(const uint2*)(pl + ln * 40 + cb + 4 * j);
            pa.w[j]     = __builtin_amdgcn_perm(dj.y, dj.x, 0x05040100u); // tile0 pair
            pa.w[j + 4] = __builtin_amdgcn_perm(dj.y, dj.x, 0x07060302u); // tile1 pair
        }

        // row sums of P via WMMA against ones
        lacc = __builtin_amdgcn_wmma_f32_16x16x32_f16(false, pa.v, false, ones.v,
                                                      (short)0, lacc, false, false);
        // P x V
        acc0 = __builtin_amdgcn_wmma_f32_16x16x32_f16(false, pa.v, false, vf0.v,
                                                      (short)0, acc0, false, false);
        acc1 = __builtin_amdgcn_wmma_f32_16x16x32_f16(false, pa.v, false, vf1.v,
                                                      (short)0, acc1, false, false);
        acc2 = __builtin_amdgcn_wmma_f32_16x16x32_f16(false, pa.v, false, vf2.v,
                                                      (short)0, acc2, false, false);
        acc3 = __builtin_amdgcn_wmma_f32_16x16x32_f16(false, pa.v, false, vf3.v,
                                                      (short)0, acc3, false, false);
    }

    // ---- epilogue: out = gamma * (acc / l) + X ----
    const float g = gamma_p[0];
    float inv[8];
#pragma unroll
    for (int i = 0; i < 8; ++i) inv[i] = 1.0f / lacc[i];
#pragma unroll
    for (int i = 0; i < 8; ++i) {
        int row = qbase + i + (hi ? 8 : 0);
        size_t o = ((size_t)b * HW + row) * NCH + ln;
        out[o + 0]  = fmaf(g, acc0[i] * inv[i], X[o + 0]);
        out[o + 16] = fmaf(g, acc1[i] * inv[i], X[o + 16]);
        out[o + 32] = fmaf(g, acc2[i] * inv[i], X[o + 32]);
        out[o + 48] = fmaf(g, acc3[i] * inv[i], X[o + 48]);
    }
}

// ---------------------------------------------------------------------------
extern "C" void kernel_launch(void* const* d_in, const int* in_sizes, int n_in,
                              void* d_out, int out_size, void* d_ws, size_t ws_size,
                              hipStream_t stream) {
    const float* X     = (const float*)d_in[0];
    const float* Wq    = (const float*)d_in[1];
    const float* bq    = (const float*)d_in[2];
    const float* Wv    = (const float*)d_in[3];
    const float* bv    = (const float*)d_in[4];
    const float* gamma = (const float*)d_in[5];
    float* out = (float*)d_out;

    const size_t QBYTES = (size_t)8 * HW * DQK * sizeof(_Float16);   // 512 KB
    _Float16* Qh = (_Float16*)d_ws;
    _Float16* Vt = (_Float16*)((char*)d_ws + QBYTES);                // 4 MB

    qv_pack_kernel<<<8 * (HW / 64), 256, 0, stream>>>(X, Wq, bq, Wv, bv, Qh, Vt);
    flash_attn_kernel<<<(8 * (HW / 16)) / 4, 128, 0, stream>>>(Qh, Vt, X, gamma, out);
}
